// Encoder_6339371729234
// MI455X (gfx1250) — matrix operand order repaired and checked by reading, not compiled, not run on hardware
//
#include <hip/hip_runtime.h>
#include <hip/hip_bf16.h>

// Problem constants (from reference)
#define L_  4
#define B_  4
#define S_  1024
#define D_  768
#define H_  12
#define DH_ 64
#define F_  3072
#define W_  128
#define R_  257   // 2*W+1

typedef _Float16 v16h __attribute__((ext_vector_type(16)));
typedef _Float16 v8h  __attribute__((ext_vector_type(8)));
typedef float    v8f  __attribute__((ext_vector_type(8)));
typedef unsigned int u32x4 __attribute__((ext_vector_type(4)));
typedef int          i32x4 __attribute__((ext_vector_type(4)));
typedef int          i32x8 __attribute__((ext_vector_type(8)));

__device__ __forceinline__ v8f wmma16(v16h a, v16h b, v8f c) {
  // D = A(16x32 f16) * B(32x16 f16) + C(16x16 f32)
  return __builtin_amdgcn_wmma_f32_16x16x32_f16(false, a, false, b, (short)0, c, false, false);
}

__device__ __forceinline__ v16h cat16(v8h lo, v8h hi) {
  v16h r;
#pragma unroll
  for (int j = 0; j < 8; ++j) { r[j] = lo[j]; r[j + 8] = hi[j]; }
  return r;
}

// ---------------------------------------------------------------------------
// TDM: DMA a 2D tile (tile_x elems wide, tile_y rows, f16) from global memory
// into LDS. tens_y clamps rows for hardware zero-fill on the N edge.
// Descriptor layout per CDNA5 ISA ch.8 (D# groups 0/1; groups 2/3 unused, 2D).
// ---------------------------------------------------------------------------
__device__ __forceinline__ void tdm_load_2d(const void* gptr, unsigned lds_addr,
                                            int tile_x, int tile_y,
                                            int tens_y, long stride_elems) {
  unsigned long long ga = (unsigned long long)gptr;
  u32x4 g0;
  g0[0] = 1u;                                        // count=1 (valid descriptor)
  g0[1] = lds_addr;                                  // LDS byte address
  g0[2] = (unsigned)ga;                              // global_addr[31:0]
  g0[3] = (unsigned)((ga >> 32) & 0x01FFFFFFull)     // global_addr[56:32]
        | (2u << 30);                                // type = 2 ("image")
  unsigned td0 = (unsigned)tile_x;                   // tensor_dim0 (no x OOB)
  unsigned td1 = (unsigned)tens_y;                   // tensor_dim1 (row clamp)
  unsigned long long st0 = (unsigned long long)stride_elems;
  i32x8 g1;
  g1[0] = (int)(1u << 16);                           // data_size=1 -> 2 bytes
  g1[1] = (int)((td0 & 0xFFFFu) << 16);              // tensor_dim0[15:0]
  g1[2] = (int)((td0 >> 16) | ((td1 & 0xFFFFu) << 16));
  g1[3] = (int)((td1 >> 16) | ((unsigned)tile_x << 16));  // tile_dim0
  g1[4] = (int)((unsigned)tile_y & 0xFFFFu);         // tile_dim1 (tile_dim2=0)
  g1[5] = (int)(unsigned)(st0 & 0xFFFFFFFFull);      // tensor_dim0_stride lo
  g1[6] = (int)(unsigned)((st0 >> 32) & 0xFFFFull);  // stride hi (stride1=0)
  g1[7] = 0;
  i32x4 z4 = {0, 0, 0, 0};
#if __clang_major__ >= 23
  i32x8 z8 = {};
  __builtin_amdgcn_tensor_load_to_lds(g0, g1, z4, z4, z8, 0);
#else
  __builtin_amdgcn_tensor_load_to_lds(g0, g1, z4, z4, 0);
#endif
}

// ---------------------------------------------------------------------------
// out[c*rows + r] = (f16) in[r*cols + c]   (batched over blockIdx.z)
// ---------------------------------------------------------------------------
__global__ __launch_bounds__(256)
void k_transpose(const float* __restrict__ in, _Float16* __restrict__ out,
                 int rows, int cols, long inStride, long outStride) {
  long z = blockIdx.z;
  const float* src = in + z * inStride;
  _Float16* dst = out + z * outStride;
  int i = blockIdx.x * 256 + threadIdx.x;
  if (i < rows * cols) {
    int r = i / cols, c = i - r * cols;
    dst[(long)c * rows + r] = (_Float16)src[(long)r * cols + c];
  }
}

// ---------------------------------------------------------------------------
// f32 -> (f32 copy, f16 copy)
// ---------------------------------------------------------------------------
__global__ __launch_bounds__(256)
void k_copy(const float* __restrict__ in, float* __restrict__ out32,
            _Float16* __restrict__ out16, long n) {
  long i = (long)blockIdx.x * 256 + threadIdx.x;
  if (i < n) {
    float v = in[i];
    out32[i] = v;
    out16[i] = (_Float16)v;
  }
}

// ---------------------------------------------------------------------------
// WMMA GEMM:  D[M,N] = A[M,K](f16, lda) * Bt[N,K](f16) (+bias +resid, gelu)
// Block tile 128x64, 8 waves (4x2), wave tile 32x32 (4 WMMAs / K-step).
// Tiles staged via the Tensor Data Mover, double-buffered: wave 0 issues the
// DMA for tile i+1 while all waves compute on tile i. M must be mult of 128.
// Batched over blockIdx.z with strides zA (A elems), zB (Bt elems), zD (D elems).
// ---------------------------------------------------------------------------
__global__ __launch_bounds__(256)
void k_gemm(const _Float16* __restrict__ A, const _Float16* __restrict__ Bt,
            const float* __restrict__ bias, const float* __restrict__ resid,
            float* __restrict__ D32, _Float16* __restrict__ D16,
            int M, int N, int K, int lda, int ldd,
            long zA, long zB, long zD, int gelu_flag) {
  __shared__ __align__(16) _Float16 As[2][128][32];   // 16 KB
  __shared__ __align__(16) _Float16 Bs[2][64][32];    //  8 KB

  long z = blockIdx.z;
  A  += z * zA;
  Bt += z * zB;
  const int m0 = blockIdx.y * 128, n0 = blockIdx.x * 64;
  const int tid = threadIdx.x;
  const int lane = tid & 31, w = tid >> 5;
  const int wy = w & 3, wx = w >> 2;       // wave tile: rows wy*32, cols wx*32
  const int m = lane & 15, g = lane >> 4;

  const unsigned ldsA0 = (unsigned)(unsigned long long)(const void*)&As[0][0][0];
  const unsigned ldsB0 = (unsigned)(unsigned long long)(const void*)&Bs[0][0][0];
  const int rowsA = M - m0;                // >= 128 except generic safety
  const int rowsB = N - n0;                // may be < 64 (N=257 edge): HW zero-fills

  v8f acc[2][2] = {};                      // [my][nb]
  const int nk = K / 32;

  if (w == 0) {
    tdm_load_2d(A + (long)m0 * lda, ldsA0, 32, 128, rowsA, lda);
    tdm_load_2d(Bt + (long)n0 * K, ldsB0, 32, 64, rowsB, K);
  }
  for (int i = 0; i < nk; ++i) {
    if (w == 0) {
      if (i + 1 < nk) {
        int buf = (i + 1) & 1;
        int k0 = (i + 1) * 32;
        tdm_load_2d(A + (long)m0 * lda + k0, ldsA0 + buf * 8192, 32, 128, rowsA, lda);
        tdm_load_2d(Bt + (long)n0 * K + k0, ldsB0 + buf * 4096, 32, 64, rowsB, K);
        __builtin_amdgcn_s_wait_tensorcnt((short)2);   // tile i complete
      } else {
        __builtin_amdgcn_s_wait_tensorcnt((short)0);
      }
    }
    __syncthreads();                        // tile i visible to all waves
    const int buf = i & 1;
#pragma unroll
    for (int my = 0; my < 2; ++my) {
      // A fragment: lane holds row m, k-chunks {g*8.., 16+g*8..}
      v16h a = cat16(*(const v8h*)&As[buf][wy * 32 + my * 16 + m][g * 8],
                     *(const v8h*)&As[buf][wy * 32 + my * 16 + m][16 + g * 8]);
#pragma unroll
      for (int nb = 0; nb < 2; ++nb) {
        // B fragment: lane holds col m, k-chunks {g*8.., 16+g*8..}
        v16h b = cat16(*(const v8h*)&Bs[buf][wx * 32 + nb * 16 + m][g * 8],
                       *(const v8h*)&Bs[buf][wx * 32 + nb * 16 + m][16 + g * 8]);
        acc[my][nb] = wmma16(a, b, acc[my][nb]);
      }
    }
    __syncthreads();                        // all done reading buf before reuse
  }

  // Epilogue. C/D frag: lane holds rows vv+8g, col m.
#pragma unroll
  for (int nb = 0; nb < 2; ++nb) {
    int col = n0 + wx * 32 + nb * 16 + m;
    if (col >= N) continue;
    float bsv = bias ? bias[col] : 0.f;
#pragma unroll
    for (int my = 0; my < 2; ++my) {
#pragma unroll
      for (int vv = 0; vv < 8; ++vv) {
        int row = m0 + wy * 32 + my * 16 + vv + 8 * g;
        if (row >= M) continue;
        float val = acc[my][nb][vv] + bsv;
        if (gelu_flag) val = 0.5f * val * (1.f + erff(val * 0.70710678118f));
        long idx = (long)row * ldd + col;
        if (resid) val += resid[idx];
        if (D32) D32[idx] = val;
        if (D16) D16[z * zD + idx] = (_Float16)val;
      }
    }
  }
}

// ---------------------------------------------------------------------------
// Banded flash attention with disentangled rel-pos bias.
// One wave per (b, h, 16-row query block). 8 waves per workgroup.
// score[i,j] = q_i.k_j + qpk[i, j-i+W] + kpq[j, W-(j-i)]  for |i-j|<=W
// ---------------------------------------------------------------------------
__global__ __launch_bounds__(256)
void k_attn(const _Float16* __restrict__ qg, const _Float16* __restrict__ kg,
            const _Float16* __restrict__ vg, const _Float16* __restrict__ qpk,
            const _Float16* __restrict__ kpq, const unsigned char* __restrict__ am,
            _Float16* __restrict__ ctx) {
  __shared__ __align__(16) _Float16 Pt[8][16][32];
  __shared__ __align__(16) _Float16 Vt[8][64][32];

  const int wid = threadIdx.x >> 5, lane = threadIdx.x & 31;
  const int task = blockIdx.x * 8 + wid;          // B*H*(S/16) tasks
  const int ib = task & 63;                       // S/16 = 64
  const int h = (task >> 6) % H_;
  const int b = task / (64 * H_);
  const int i0 = ib * 16;
  const int m = lane & 15, g = lane >> 4;

  // Q fragments (rows i0..i0+15, K-dim 64 -> two K=32 frags)
  const _Float16* qrow = qg + ((long)b * S_ + i0 + m) * D_ + h * DH_;
  v16h a0 = cat16(*(const v8h*)(qrow + g * 8), *(const v8h*)(qrow + 16 + g * 8));
  v16h a1 = cat16(*(const v8h*)(qrow + 32 + g * 8), *(const v8h*)(qrow + 48 + g * 8));

  bool imask[8];
#pragma unroll
  for (int vv = 0; vv < 8; ++vv) imask[vv] = am[b * S_ + i0 + vv + 8 * g] != 0;

  v8f O[4] = {};
  float mrow[8], lsum[8];
#pragma unroll
  for (int vv = 0; vv < 8; ++vv) { mrow[vv] = -1e30f; lsum[vv] = 0.f; }

  auto ldk = [&](int jbase, int kk) -> v16h {
    int row = jbase + m;    // key index for this lane's column
    if (row < S_) {
      const _Float16* p = kg + ((long)b * S_ + row) * D_ + h * DH_ + kk * 32 + g * 8;
      return cat16(*(const v8h*)p, *(const v8h*)(p + 16));
    }
    v16h zz = {};
    return zz;
  };

  const int js = (i0 > W_) ? ((i0 - W_) & ~31) : 0;
  const int je0 = i0 + W_ + 16;
  const int je = (je0 < S_) ? je0 : S_;
  const long hb = ((long)(b * H_ + h)) * S_;

  for (int j0 = js; j0 < je; j0 += 32) {
    // --- scores: 16 x 32 chunk, as two 16x16 f32 frags ---
    v8f sc[2];
#pragma unroll
    for (int jj = 0; jj < 2; ++jj) {
      v8f acc = {};
      acc = wmma16(a0, ldk(j0 + 16 * jj, 0), acc);
      acc = wmma16(a1, ldk(j0 + 16 * jj, 1), acc);
      sc[jj] = acc;
    }
    // --- rel-pos bias + band/seq mask ---
    float rmax[8];
#pragma unroll
    for (int vv = 0; vv < 8; ++vv) rmax[vv] = -1e30f;
#pragma unroll
    for (int jj = 0; jj < 2; ++jj) {
      int j = j0 + 16 * jj + m;
      bool jok = (j < S_) && (am[b * S_ + j] != 0);
#pragma unroll
      for (int vv = 0; vv < 8; ++vv) {
        int i = i0 + vv + 8 * g;
        int d = j - i;
        float val = -1e30f;
        if (jok && imask[vv] && d >= -W_ && d <= W_) {
          val = sc[jj][vv] + (float)qpk[(hb + i) * R_ + (d + W_)]
                          + (float)kpq[(hb + j) * R_ + (W_ - d)];
        }
        sc[jj][vv] = val;
        rmax[vv] = fmaxf(rmax[vv], val);
      }
    }
    // row max across the 16 lanes of this half-wave
#pragma unroll
    for (int off = 1; off < 16; off <<= 1)
#pragma unroll
      for (int vv = 0; vv < 8; ++vv)
        rmax[vv] = fmaxf(rmax[vv], __shfl_xor(rmax[vv], off, 32));

    float scale[8], rsum[8];
#pragma unroll
    for (int vv = 0; vv < 8; ++vv) {
      float mn = fmaxf(mrow[vv], rmax[vv]);
      scale[vv] = expf(mrow[vv] - mn);
      mrow[vv] = mn;
      rsum[vv] = 0.f;
    }
    // P = exp(s - m), store to this wave's LDS P tile
#pragma unroll
    for (int jj = 0; jj < 2; ++jj)
#pragma unroll
      for (int vv = 0; vv < 8; ++vv) {
        float pv = (sc[jj][vv] > -1e29f) ? expf(sc[jj][vv] - mrow[vv]) : 0.f;
        rsum[vv] += pv;
        Pt[wid][vv + 8 * g][16 * jj + m] = (_Float16)pv;
      }
#pragma unroll
    for (int off = 1; off < 16; off <<= 1)
#pragma unroll
      for (int vv = 0; vv < 8; ++vv) rsum[vv] += __shfl_xor(rsum[vv], off, 32);
#pragma unroll
    for (int vv = 0; vv < 8; ++vv) lsum[vv] = lsum[vv] * scale[vv] + rsum[vv];
#pragma unroll
    for (int nb = 0; nb < 4; ++nb)
#pragma unroll
      for (int vv = 0; vv < 8; ++vv) O[nb][vv] *= scale[vv];

    // V tile (32 keys x 64 dims) -> LDS transposed Vt[c][k]; vector global loads
#pragma unroll
    for (int t = 0; t < 8; ++t) {
      int idx = t * 32 + lane;          // 256 chunks of 8 halves
      int r = idx >> 3;                 // key row 0..31
      int c = (idx & 7) * 8;            // col group
      v8h val = {};
      int row = j0 + r;
      if (row < S_) val = *(const v8h*)(vg + ((long)b * S_ + row) * D_ + h * DH_ + c);
#pragma unroll
      for (int e = 0; e < 8; ++e) Vt[wid][c + e][r] = val[e];
    }
    asm volatile("s_wait_dscnt 0" ::: "memory");   // intra-wave LDS ordering

    // P as A-frag (16x32), V as four B-frags (32x16 each) -> O += P*V
    v16h pa = cat16(*(const v8h*)&Pt[wid][m][g * 8],
                    *(const v8h*)&Pt[wid][m][16 + g * 8]);
#pragma unroll
    for (int nb = 0; nb < 4; ++nb) {
      v16h vb = cat16(*(const v8h*)&Vt[wid][nb * 16 + m][g * 8],
                      *(const v8h*)&Vt[wid][nb * 16 + m][16 + g * 8]);
      O[nb] = wmma16(pa, vb, O[nb]);
    }
  }

  // normalize and write context (B,S,D) f16
#pragma unroll
  for (int nb = 0; nb < 4; ++nb)
#pragma unroll
    for (int vv = 0; vv < 8; ++vv) {
      int i = i0 + vv + 8 * g;
      float den = lsum[vv];
      float val = (den > 1e-30f) ? O[nb][vv] / den : 0.f;
      ctx[((long)b * S_ + i) * D_ + h * DH_ + nb * 16 + m] = (_Float16)val;
    }
}

// ---------------------------------------------------------------------------
// LayerNorm over last dim (768), one block per row; writes f32 + f16 copies.
// ---------------------------------------------------------------------------
__global__ __launch_bounds__(256)
void k_ln(const float* __restrict__ in, const float* __restrict__ gam,
          const float* __restrict__ bet, float* __restrict__ out32,
          _Float16* __restrict__ out16) {
  __shared__ float r1[256], r2[256];
  long row = blockIdx.x;
  const float* x = in + row * D_;
  int t = threadIdx.x;
  float s = 0.f, s2 = 0.f;
  for (int i = t; i < D_; i += 256) { float v = x[i]; s += v; s2 += v * v; }
  r1[t] = s; r2[t] = s2;
  __syncthreads();
  for (int off = 128; off > 0; off >>= 1) {
    if (t < off) { r1[t] += r1[t + off]; r2[t] += r2[t + off]; }
    __syncthreads();
  }
  float mu = r1[0] * (1.f / D_);
  float var = r2[0] * (1.f / D_) - mu * mu;
  float inv = rsqrtf(var + 1e-8f);
  for (int i = t; i < D_; i += 256) {
    float v = (x[i] - mu) * inv * gam[i] + bet[i];
    out32[row * D_ + i] = v;
    out16[row * D_ + i] = (_Float16)v;
  }
}

// ---------------------------------------------------------------------------
extern "C" void kernel_launch(void* const* d_in, const int* in_sizes, int n_in,
                              void* d_out, int out_size, void* d_ws, size_t ws_size,
                              hipStream_t stream) {
  (void)in_sizes; (void)n_in; (void)out_size; (void)ws_size;
  const float* hs = (const float*)d_in[0];
  const unsigned char* am = (const unsigned char*)d_in[2];
  const float* wq = (const float*)d_in[3];
  const float* wk = (const float*)d_in[4];
  const float* wv = (const float*)d_in[5];
  const float* pq = (const float*)d_in[6];
  const float* pk = (const float*)d_in[7];
  const float* wo = (const float*)d_in[8];
  const float* bo = (const float*)d_in[9];
  const float* w1 = (const float*)d_in[10];
  const float* b1 = (const float*)d_in[11];
  const float* w2 = (const float*)d_in[12];
  const float* b2 = (const float*)d_in[13];
  const float* lg = (const float*)d_in[14];
  const float* lb = (const float*)d_in[15];

  char* p = (char*)d_ws;
  auto carve = [&](size_t bytes) -> void* {
    char* r = p;
    p += (bytes + 255) & ~(size_t)255;
    return (void*)r;
  };
  const size_t BSD = (size_t)B_ * S_ * D_;
  float* h32 = (float*)carve(BSD * 4);
  float* hn32 = (float*)carve(BSD * 4);
  float* tmp32 = (float*)carve(BSD * 4);
  _Float16* h16 = (_Float16*)carve(BSD * 2);
  _Float16* hn16 = (_Float16*)carve(BSD * 2);
  _Float16* q16 = (_Float16*)carve(BSD * 2);
  _Float16* k16 = (_Float16*)carve(BSD * 2);
  _Float16* v16 = (_Float16*)carve(BSD * 2);
  _Float16* ctx16 = (_Float16*)carve(BSD * 2);
  _Float16* inter16 = (_Float16*)carve((size_t)B_ * S_ * F_ * 2);
  _Float16* qpk16 = (_Float16*)carve((size_t)B_ * H_ * S_ * R_ * 2);
  _Float16* kpq16 = (_Float16*)carve((size_t)B_ * H_ * S_ * R_ * 2);
  _Float16* wqT = (_Float16*)carve((size_t)D_ * D_ * 2);
  _Float16* wkT = (_Float16*)carve((size_t)D_ * D_ * 2);
  _Float16* wvT = (_Float16*)carve((size_t)D_ * D_ * 2);
  _Float16* woT = (_Float16*)carve((size_t)D_ * D_ * 2);
  _Float16* w1T = (_Float16*)carve((size_t)D_ * F_ * 2);
  _Float16* w2T = (_Float16*)carve((size_t)D_ * F_ * 2);
  _Float16* pqT = (_Float16*)carve((size_t)H_ * R_ * DH_ * 2);
  _Float16* pkT = (_Float16*)carve((size_t)H_ * R_ * DH_ * 2);

  k_copy<<<dim3((unsigned)((BSD + 255) / 256)), 256, 0, stream>>>(hs, h32, h16, (long)BSD);

  for (int l = 0; l < L_; ++l) {
    // --- weight transposes (f32 -> f16, N x K layout) ---
    dim3 tDD((D_ * D_ + 255) / 256, 1, 1);
    k_transpose<<<tDD, 256, 0, stream>>>(wq + (size_t)l * D_ * D_, wqT, D_, D_, 0, 0);
    k_transpose<<<tDD, 256, 0, stream>>>(wk + (size_t)l * D_ * D_, wkT, D_, D_, 0, 0);
    k_transpose<<<tDD, 256, 0, stream>>>(wv + (size_t)l * D_ * D_, wvT, D_, D_, 0, 0);
    k_transpose<<<tDD, 256, 0, stream>>>(wo + (size_t)l * D_ * D_, woT, D_, D_, 0, 0);
    dim3 tDF((D_ * F_ + 255) / 256, 1, 1);
    k_transpose<<<tDF, 256, 0, stream>>>(w1 + (size_t)l * D_ * F_, w1T, D_, F_, 0, 0);
    k_transpose<<<tDF, 256, 0, stream>>>(w2 + (size_t)l * F_ * D_, w2T, F_, D_, 0, 0);
    dim3 tP((DH_ * R_ + 255) / 256, 1, H_);
    k_transpose<<<tP, 256, 0, stream>>>(pq + (size_t)l * H_ * DH_ * R_, pqT, DH_, R_,
                                        (long)DH_ * R_, (long)R_ * DH_);
    k_transpose<<<tP, 256, 0, stream>>>(pk + (size_t)l * H_ * DH_ * R_, pkT, DH_, R_,
                                        (long)DH_ * R_, (long)R_ * DH_);

    // --- QKV projections (f16 out) ---
    dim3 gD(D_ / 64, (B_ * S_) / 128, 1);
    k_gemm<<<gD, 256, 0, stream>>>(h16, wqT, nullptr, nullptr, nullptr, q16,
                                   B_ * S_, D_, D_, D_, D_, 0, 0, 0, 0);
    k_gemm<<<gD, 256, 0, stream>>>(h16, wkT, nullptr, nullptr, nullptr, k16,
                                   B_ * S_, D_, D_, D_, D_, 0, 0, 0, 0);
    k_gemm<<<gD, 256, 0, stream>>>(h16, wvT, nullptr, nullptr, nullptr, v16,
                                   B_ * S_, D_, D_, D_, D_, 0, 0, 0, 0);

    // --- position strips: qpk = q.pos_k, kpq = k.pos_q  (batched over heads) ---
    dim3 gP((R_ + 63) / 64, S_ / 128, H_);
    for (int b = 0; b < B_; ++b) {
      k_gemm<<<gP, 256, 0, stream>>>(q16 + (size_t)b * S_ * D_, pkT,
                                     nullptr, nullptr, nullptr,
                                     qpk16 + (size_t)b * H_ * S_ * R_,
                                     S_, R_, DH_, D_, R_,
                                     (long)DH_, (long)R_ * DH_, (long)S_ * R_, 0);
      k_gemm<<<gP, 256, 0, stream>>>(k16 + (size_t)b * S_ * D_, pqT,
                                     nullptr, nullptr, nullptr,
                                     kpq16 + (size_t)b * H_ * S_ * R_,
                                     S_, R_, DH_, D_, R_,
                                     (long)DH_, (long)R_ * DH_, (long)S_ * R_, 0);
    }

    // --- banded flash attention ---
    k_attn<<<dim3(B_ * H_ * (S_ / 16) / 8), 256, 0, stream>>>(q16, k16, v16, qpk16,
                                                              kpq16, am, ctx16);

    // --- output projection + residual ---
    k_gemm<<<gD, 256, 0, stream>>>(ctx16, woT, bo + (size_t)l * D_, h32, hn32, hn16,
                                   B_ * S_, D_, D_, D_, D_, 0, 0, 0, 0);

    // --- FFN ---
    dim3 gF(F_ / 64, (B_ * S_) / 128, 1);
    k_gemm<<<gF, 256, 0, stream>>>(hn16, w1T, b1 + (size_t)l * F_, nullptr, nullptr,
                                   inter16, B_ * S_, F_, D_, D_, F_, 0, 0, 0, 1);
    k_gemm<<<gD, 256, 0, stream>>>(inter16, w2T, b2 + (size_t)l * D_, hn32, tmp32,
                                   nullptr, B_ * S_, D_, F_, F_, D_, 0, 0, 0, 0);

    // --- LayerNorm ---
    float* lnout = (l == L_ - 1) ? (float*)d_out : h32;
    k_ln<<<dim3(B_ * S_), 256, 0, stream>>>(tmp32, lg + (size_t)l * D_,
                                            lb + (size_t)l * D_, lnout, h16);
  }
}
